// Attention_17008070493108
// MI455X (gfx1250) — matrix operand order repaired and checked by reading, not compiled, not run on hardware
//
#include <hip/hip_runtime.h>

// ---------------------------------------------------------------------------
// Types
// ---------------------------------------------------------------------------
typedef __bf16         bf16x16  __attribute__((ext_vector_type(16)));
typedef float          f32x8    __attribute__((ext_vector_type(8)));
typedef unsigned int   uint32x4 __attribute__((ext_vector_type(4)));
typedef unsigned int   v4u      __attribute__((ext_vector_type(4)));
typedef int            v8i      __attribute__((ext_vector_type(8)));
typedef int            v4i      __attribute__((ext_vector_type(4)));
typedef unsigned short u16;

union Frag { bf16x16 v; uint32x4 q[2]; };

// Problem constants (from reference)
constexpr int BB   = 2;
constexpr int NN   = 2048;
constexpr int DIM  = 1024;
constexpr int HH   = 16;
constexpr int HD   = 64;
constexpr int MROW = BB * NN;          // 4096 token rows
constexpr float EPS = 1e-6f;

// f32 -> bf16 (round to nearest even)
__device__ __forceinline__ u16 f2bf(float f) {
    unsigned u = __builtin_bit_cast(unsigned, f);
    u += 0x7fffu + ((u >> 16) & 1u);
    return (u16)(u >> 16);
}

// ds_swizzle xor-shuffle within wave32 (and_mask=0x1f, or=0, xor=m)
#define SWZ(x, m) __builtin_bit_cast(float, \
    __builtin_amdgcn_ds_swizzle(__builtin_bit_cast(int, (x)), (((m) << 10) | 0x1f)))

#define WMMA_BF16(a, b, c) \
    __builtin_amdgcn_wmma_f32_16x16x32_bf16(false, (a), false, (b), (short)0, (c), false, false)

// ---------------------------------------------------------------------------
// Tensor Data Mover: 2-D tile (bf16 elements) global -> LDS.
// D# packing per CDNA5 ISA 8.3/8.4 (group0: count/lds/global/type,
// group1: data_size=2B, dims, tiles, strides). Groups 2/3 zero (<=2D tensor).
// 6-arg builtin (clang-23 / therock): (v4u, v8i, v4i, v4i, v8i, i32 cpol).
// ---------------------------------------------------------------------------
__device__ __forceinline__ void tdm_load_2d_bf16(unsigned lds_addr, const void* gptr,
                                                 unsigned dim0, unsigned dim1,
                                                 unsigned tile0, unsigned tile1,
                                                 unsigned long long stride0) {
    unsigned long long ga = (unsigned long long)(size_t)gptr;
    v4u g0;
    g0[0] = 1u;                                              // count=1 (valid user D#)
    g0[1] = lds_addr;                                        // LDS byte address
    g0[2] = (unsigned)ga;                                    // global_addr[31:0]
    g0[3] = (unsigned)((ga >> 32) & 0x01FFFFFFu) | (2u << 30); // addr[56:32] | type=2
    v8i g1;
    g1[0] = (int)(1u << 16);                                 // wg_mask=0, data_size=1 (2B)
    g1[1] = (int)((dim0 & 0xFFFFu) << 16);                   // tensor_dim0[15:0]
    g1[2] = (int)(((dim0 >> 16) & 0xFFFFu) | ((dim1 & 0xFFFFu) << 16));
    g1[3] = (int)(((dim1 >> 16) & 0xFFFFu) | ((tile0 & 0xFFFFu) << 16)); // tile_dim0
    g1[4] = (int)(tile1 & 0xFFFFu);                          // tile_dim1, tile_dim2=0
    g1[5] = (int)(unsigned)(stride0 & 0xFFFFFFFFu);          // tensor_dim0_stride lo
    g1[6] = (int)(unsigned)((stride0 >> 32) & 0xFFFFu);      // stride0 hi, stride1=0
    g1[7] = 0;
    v4i z4 = {0, 0, 0, 0};
    v8i z8 = {0, 0, 0, 0, 0, 0, 0, 0};
    __builtin_amdgcn_tensor_load_to_lds(g0, g1, z4, z4, z8, 0);
}

// ---------------------------------------------------------------------------
// 1) f32 -> bf16 flat convert
// ---------------------------------------------------------------------------
__global__ __launch_bounds__(256) void cvt_bf16_kernel(const float* __restrict__ src,
                                                       u16* __restrict__ dst, int n) {
    int i = blockIdx.x * 256 + threadIdx.x;
    if (i < n) dst[i] = f2bf(src[i]);
}

// ---------------------------------------------------------------------------
// 2) W [K,N] f32 -> Wt [N,K] bf16 (so B-fragments read k-contiguous)
// ---------------------------------------------------------------------------
__global__ __launch_bounds__(256) void transpose_w_kernel(const float* __restrict__ W,
                                                          u16* __restrict__ Wt,
                                                          int K, int N) {
    int idx = blockIdx.x * 256 + threadIdx.x;
    if (idx >= K * N) return;
    int k = idx / N, n = idx % N;                 // coalesced read of W
    Wt[(size_t)n * K + k] = f2bf(W[idx]);
}

// ---------------------------------------------------------------------------
// 3) GEMM: C[M,N] = A[M,K](bf16) * Bt[N,K]^T(bf16) + bias   (f32 accum)
//    one wave computes a 16x64 tile (4 WMMA accumulators, A fragment reused 4x).
//    All 4 B fragments are loaded into disjoint registers BEFORE the WMMAs so
//    the loads clause together and the WMMAs drain with partial loadcnt waits.
// ---------------------------------------------------------------------------
__global__ __launch_bounds__(256) void gemm_bf16_kernel(const u16* __restrict__ A,
                                                        const u16* __restrict__ Bt,
                                                        const float* __restrict__ bias,
                                                        float* __restrict__ C,
                                                        int M, int N, int K) {
    const int wid  = threadIdx.x >> 5;
    const int lane = threadIdx.x & 31;
    const int wave = blockIdx.x * 8 + wid;
    const int tilesN = N >> 6;                       // 64-col tiles
    const int tm = wave / tilesN, tn = wave % tilesN;
    if (tm * 16 >= M) return;

    const int half = lane >> 4, l16 = lane & 15;
    const int kbA = half * 8;      // A frag: lanes 0-15 -> K 0..7/16..23, 16-31 -> 8..15/24..31
    const int kbB = half * 8;      // B frag: lanes 0-15 -> K 0..7/16..23, 16-31 -> 8..15/24..31

    const u16* Arow = A + (size_t)(tm * 16 + l16) * K;
    const u16* Brow0 = Bt + (size_t)(tn * 64 + l16) * K;

    f32x8 acc[4];
#pragma unroll
    for (int j = 0; j < 4; ++j) acc[j] = f32x8{0.f, 0.f, 0.f, 0.f, 0.f, 0.f, 0.f, 0.f};

#pragma unroll 2
    for (int k0 = 0; k0 < K; k0 += 32) {
        Frag a, b[4];
        a.q[0] = *(const uint32x4*)(Arow + k0 + kbA);
        a.q[1] = *(const uint32x4*)(Arow + k0 + kbA + 16);
#pragma unroll
        for (int j = 0; j < 4; ++j) {
            const u16* Brow = Brow0 + (size_t)(j * 16) * K;
            b[j].q[0] = *(const uint32x4*)(Brow + k0 + kbB);
            b[j].q[1] = *(const uint32x4*)(Brow + k0 + kbB + 16);
        }
#pragma unroll
        for (int j = 0; j < 4; ++j)
            acc[j] = WMMA_BF16(a.v, b[j].v, acc[j]);
    }
#pragma unroll
    for (int j = 0; j < 4; ++j) {
        const int n = tn * 64 + j * 16 + l16;
        const float bv = bias[n];
#pragma unroll
        for (int r = 0; r < 8; ++r) {
            const int row = tm * 16 + r + 8 * half;  // C layout: VGPR r -> row r (+8 hi lanes)
            C[(size_t)row * N + n] = acc[j][r] + bv;
        }
    }
}

// ---------------------------------------------------------------------------
// 4) Per-head layernorm + RoPE; pack to WMMA-friendly bf16 layouts.
//    One wave per (b,n,h); lane holds d=lane and d=lane+32.
//    Qn,Kn: [B,H,N,HD] bf16;   VT: [B,H,HD,N] bf16 (transposed for PV B-frags)
// ---------------------------------------------------------------------------
__global__ __launch_bounds__(256) void norm_rope_kernel(
    const float* __restrict__ Qr, const float* __restrict__ Kr, const float* __restrict__ Vr,
    const float* __restrict__ rcos, const float* __restrict__ rsin,
    const float* __restrict__ qnw, const float* __restrict__ qnb,
    const float* __restrict__ knw, const float* __restrict__ knb,
    u16* __restrict__ Qn, u16* __restrict__ Kn, u16* __restrict__ VT) {
    const int wave = blockIdx.x * 8 + (threadIdx.x >> 5);
    const int lane = threadIdx.x & 31;
    if (wave >= MROW * HH) return;
    const int h  = wave % HH;
    const int bn = wave / HH;                // b*N + n
    const int b  = bn / NN, n = bn % NN;

    const size_t roff = (size_t)bn * DIM + h * HD;
    const float c0 = rcos[(size_t)bn * HD + lane];
    const float c1 = rcos[(size_t)bn * HD + lane + 32];
    const float s0 = rsin[(size_t)bn * HD + lane];
    const float s1 = rsin[(size_t)bn * HD + lane + 32];

    const size_t qko = ((size_t)(b * HH + h) * NN + n) * HD;
    const size_t vto = (size_t)(b * HH + h) * HD * NN + n;

    // ---- Q: layernorm + rope ----
    {
        float x0 = Qr[roff + lane], x1 = Qr[roff + lane + 32];
        float s = x0 + x1, sq = x0 * x0 + x1 * x1;
        s  += SWZ(s, 1);  s  += SWZ(s, 2);  s  += SWZ(s, 4);  s  += SWZ(s, 8);  s  += SWZ(s, 16);
        sq += SWZ(sq, 1); sq += SWZ(sq, 2); sq += SWZ(sq, 4); sq += SWZ(sq, 8); sq += SWZ(sq, 16);
        const float mu  = s * (1.f / 64.f);
        const float var = sq * (1.f / 64.f) - mu * mu;
        const float rs  = rsqrtf(var + EPS);
        float y0 = (x0 - mu) * rs * qnw[lane]      + qnb[lane];
        float y1 = (x1 - mu) * rs * qnw[lane + 32] + qnb[lane + 32];
        Qn[qko + lane]      = f2bf(y0 * c0 - y1 * s0);   // rot[d]   = -x[d+32]
        Qn[qko + lane + 32] = f2bf(y1 * c1 + y0 * s1);   // rot[d+32]=  x[d]
    }
    // ---- K: layernorm + rope ----
    {
        float x0 = Kr[roff + lane], x1 = Kr[roff + lane + 32];
        float s = x0 + x1, sq = x0 * x0 + x1 * x1;
        s  += SWZ(s, 1);  s  += SWZ(s, 2);  s  += SWZ(s, 4);  s  += SWZ(s, 8);  s  += SWZ(s, 16);
        sq += SWZ(sq, 1); sq += SWZ(sq, 2); sq += SWZ(sq, 4); sq += SWZ(sq, 8); sq += SWZ(sq, 16);
        const float mu  = s * (1.f / 64.f);
        const float var = sq * (1.f / 64.f) - mu * mu;
        const float rs  = rsqrtf(var + EPS);
        float y0 = (x0 - mu) * rs * knw[lane]      + knb[lane];
        float y1 = (x1 - mu) * rs * knw[lane + 32] + knb[lane + 32];
        Kn[qko + lane]      = f2bf(y0 * c0 - y1 * s0);
        Kn[qko + lane + 32] = f2bf(y1 * c1 + y0 * s1);
    }
    // ---- V: transpose to [HD,N] ----
    {
        VT[vto + (size_t)lane * NN]        = f2bf(Vr[roff + lane]);
        VT[vto + (size_t)(lane + 32) * NN] = f2bf(Vr[roff + lane + 32]);
    }
}

// ---------------------------------------------------------------------------
// 5) Flash attention. Block = 8 waves, same (b,h); waves own adjacent 16-query
//    tiles. K/V tiles for each 32-key block are DMA'd global->LDS by the
//    Tensor Data Mover (wave 0, double-buffered, s_wait_tensorcnt + barrier);
//    all waves consume fragments with ds_load_b128.
//    S = QK^T via 4 WMMA, online softmax (ds_swizzle row-max),
//    P staged bf16 through LDS (C->A layout), O += P*V^T via 4 WMMA.
// ---------------------------------------------------------------------------
__global__ __launch_bounds__(256) void flash_attn_kernel(const u16* __restrict__ Qn,
                                                         const u16* __restrict__ Kn,
                                                         const u16* __restrict__ VT,
                                                         u16* __restrict__ Obf) {
    // [0,2048)x2 : K tiles (32 keys x 64 d)   double buffered
    // [4096,..)x2: V^T tiles (64 d x 32 keys) double buffered
    // [8192,..)  : per-wave P staging (16x32)
    __shared__ u16 smem[12288];
    const int wid  = threadIdx.x >> 5;
    const int lane = threadIdx.x & 31;
    const int wave = blockIdx.x * 8 + wid;
    const int qt = wave % (NN / 16);
    const int bh = wave / (NN / 16);                 // all 8 waves share bh
    const int b = bh / HH, h = bh % HH;
    const int half = lane >> 4, l16 = lane & 15;
    const int kbA = half * 8, kbB = half * 8;
    const float scale = 0.125f;                      // 1/sqrt(64)

    const u16* Kbase = Kn + (size_t)bh * NN * HD;
    const u16* Vbase = VT + (size_t)bh * HD * NN;
    u16* P = smem + 8192 + wid * 512;

    // Q A-fragments (d 0..31 and 32..63), loaded once from global
    const u16* Qrow = Qn + ((size_t)bh * NN + qt * 16 + l16) * HD;
    Frag qa0, qa1;
    qa0.q[0] = *(const uint32x4*)(Qrow + kbA);
    qa0.q[1] = *(const uint32x4*)(Qrow + kbA + 16);
    qa1.q[0] = *(const uint32x4*)(Qrow + 32 + kbA);
    qa1.q[1] = *(const uint32x4*)(Qrow + 32 + kbA + 16);

    auto tdm_issue = [&](int ibuf, int kb) {
        // K tile: 32 contiguous rows of 64 bf16 -> flat 2048-element line
        tdm_load_2d_bf16((unsigned)(size_t)(smem + ibuf * 2048), Kbase + (size_t)kb * HD,
                         /*dim0*/ 32 * HD, /*dim1*/ 1, /*tile0*/ 32 * HD, /*tile1*/ 1,
                         /*stride0*/ 32 * HD);
        // V^T tile: 64 rows (hd), 32 cols starting at key kb, row stride N
        tdm_load_2d_bf16((unsigned)(size_t)(smem + 4096 + ibuf * 2048), Vbase + kb,
                         /*dim0*/ NN, /*dim1*/ HD, /*tile0*/ 32, /*tile1*/ HD,
                         /*stride0*/ NN);
    };
    if (wid == 0) tdm_issue(0, 0);

    f32x8 o[4];
    float mrow[8], lrow[8];
#pragma unroll
    for (int t = 0; t < 4; ++t) o[t] = f32x8{0.f, 0.f, 0.f, 0.f, 0.f, 0.f, 0.f, 0.f};
#pragma unroll
    for (int r = 0; r < 8; ++r) { mrow[r] = -3.0e38f; lrow[r] = 0.f; }

    for (int it = 0; it < NN / 32; ++it) {
        const int kb = it * 32;
        const int ibuf = it & 1;
        if (wid == 0) __builtin_amdgcn_s_wait_tensorcnt(0);   // current buffers landed
        __syncthreads();                                      // publish to all waves
        if (wid == 0 && it + 1 < NN / 32) tdm_issue(ibuf ^ 1, kb + 32);

        const u16* Kt = smem + ibuf * 2048;
        const u16* Vt = smem + 4096 + ibuf * 2048;

        // ---- S tiles from LDS: keys 0..15 and 16..31 of this block ----
        const u16* Kr0 = Kt + (size_t)l16 * HD;
        const u16* Kr1 = Kt + (size_t)(16 + l16) * HD;
        Frag k00, k01, k10, k11;
        k00.q[0] = *(const uint32x4*)(Kr0 + kbB);       k00.q[1] = *(const uint32x4*)(Kr0 + kbB + 16);
        k01.q[0] = *(const uint32x4*)(Kr0 + 32 + kbB);  k01.q[1] = *(const uint32x4*)(Kr0 + 32 + kbB + 16);
        k10.q[0] = *(const uint32x4*)(Kr1 + kbB);       k10.q[1] = *(const uint32x4*)(Kr1 + kbB + 16);
        k11.q[0] = *(const uint32x4*)(Kr1 + 32 + kbB);  k11.q[1] = *(const uint32x4*)(Kr1 + 32 + kbB + 16);

        f32x8 s0 = {0.f, 0.f, 0.f, 0.f, 0.f, 0.f, 0.f, 0.f};
        f32x8 s1 = {0.f, 0.f, 0.f, 0.f, 0.f, 0.f, 0.f, 0.f};
        s0 = WMMA_BF16(qa0.v, k00.v, s0);
        s0 = WMMA_BF16(qa1.v, k01.v, s0);
        s1 = WMMA_BF16(qa0.v, k10.v, s1);
        s1 = WMMA_BF16(qa1.v, k11.v, s1);

        // ---- online softmax update + stage P to LDS (bf16, A-layout rows) ----
#pragma unroll
        for (int r = 0; r < 8; ++r) {
            const float a = s0[r] * scale, c = s1[r] * scale;
            float mx = fmaxf(a, c);
            mx = fmaxf(mx, SWZ(mx, 1)); mx = fmaxf(mx, SWZ(mx, 2));
            mx = fmaxf(mx, SWZ(mx, 4)); mx = fmaxf(mx, SWZ(mx, 8));
            const float mnew = fmaxf(mrow[r], mx);
            const float corr = __expf(mrow[r] - mnew);
            mrow[r] = mnew;
            const float p0 = __expf(a - mnew);
            const float p1 = __expf(c - mnew);
            lrow[r] = lrow[r] * corr + p0 + p1;
            o[0][r] *= corr; o[1][r] *= corr; o[2][r] *= corr; o[3][r] *= corr;
            const int row = r + 8 * half;
            P[row * 32 + l16]      = f2bf(p0);
            P[row * 32 + 16 + l16] = f2bf(p1);
        }
        asm volatile("s_wait_dscnt 0" ::: "memory");

        // ---- P A-fragment (16x32) and V^T B-fragments from LDS; O += P*V^T ----
        Frag pa;
        pa.q[0] = *(const uint32x4*)(P + l16 * 32 + kbA);
        pa.q[1] = *(const uint32x4*)(P + l16 * 32 + kbA + 16);
        Frag vb[4];
#pragma unroll
        for (int t = 0; t < 4; ++t) {
            const u16* Vrow = Vt + (size_t)(t * 16 + l16) * 32;
            vb[t].q[0] = *(const uint32x4*)(Vrow + kbB);
            vb[t].q[1] = *(const uint32x4*)(Vrow + kbB + 16);
        }
#pragma unroll
        for (int t = 0; t < 4; ++t)
            o[t] = WMMA_BF16(pa.v, vb[t].v, o[t]);
    }

    // ---- normalize and write O as bf16 [B*N, DIM] for the output GEMM ----
#pragma unroll
    for (int r = 0; r < 8; ++r) {
        float lt = lrow[r];
        lt += SWZ(lt, 1); lt += SWZ(lt, 2); lt += SWZ(lt, 4); lt += SWZ(lt, 8);
        const float inv = 1.f / lt;
        const int row = b * NN + qt * 16 + r + 8 * half;
#pragma unroll
        for (int t = 0; t < 4; ++t) {
            const int col = h * HD + t * 16 + l16;
            Obf[(size_t)row * DIM + col] = f2bf(o[t][r] * inv);
        }
    }
}

// ---------------------------------------------------------------------------
// Host-side orchestration
// ---------------------------------------------------------------------------
extern "C" void kernel_launch(void* const* d_in, const int* in_sizes, int n_in,
                              void* d_out, int out_size, void* d_ws, size_t ws_size,
                              hipStream_t stream) {
    const float* x    = (const float*)d_in[0];
    const float* rcos = (const float*)d_in[1];
    const float* rsin = (const float*)d_in[2];
    const float* Wq = (const float*)d_in[3];  const float* bq = (const float*)d_in[4];
    const float* Wk = (const float*)d_in[5];  const float* bk = (const float*)d_in[6];
    const float* Wv = (const float*)d_in[7];  const float* bv = (const float*)d_in[8];
    const float* qnw = (const float*)d_in[9];  const float* qnb = (const float*)d_in[10];
    const float* knw = (const float*)d_in[11]; const float* knb = (const float*)d_in[12];
    const float* Wo = (const float*)d_in[13]; const float* bo = (const float*)d_in[14];
    float* out = (float*)d_out;

    const size_t M = MROW, K = DIM, N = DIM;

    char* p = (char*)d_ws;
    auto carve = [&](size_t bytes) { char* r = p; p += (bytes + 255) & ~(size_t)255; return (void*)r; };
    u16*   Xb   = (u16*)  carve(M * K * sizeof(u16));
    u16*   WqT  = (u16*)  carve(K * N * sizeof(u16));
    u16*   WkT  = (u16*)  carve(K * N * sizeof(u16));
    u16*   WvT  = (u16*)  carve(K * N * sizeof(u16));
    u16*   WoT  = (u16*)  carve(K * N * sizeof(u16));
    float* Qraw = (float*)carve(M * N * sizeof(float));
    float* Kraw = (float*)carve(M * N * sizeof(float));
    float* Vraw = (float*)carve(M * N * sizeof(float));
    u16*   Qn   = (u16*)  carve((size_t)BB * HH * NN * HD * sizeof(u16));
    u16*   Kn   = (u16*)  carve((size_t)BB * HH * NN * HD * sizeof(u16));
    u16*   VTb  = (u16*)  carve((size_t)BB * HH * HD * NN * sizeof(u16));
    u16*   Obf  = (u16*)  carve(M * DIM * sizeof(u16));

    // 1) x -> bf16
    cvt_bf16_kernel<<<(int)((M * K) / 256), 256, 0, stream>>>(x, Xb, (int)(M * K));
    // 2) weight transposes -> bf16 [N,K]
    const int tw_blocks = (int)((K * N) / 256);
    transpose_w_kernel<<<tw_blocks, 256, 0, stream>>>(Wq, WqT, (int)K, (int)N);
    transpose_w_kernel<<<tw_blocks, 256, 0, stream>>>(Wk, WkT, (int)K, (int)N);
    transpose_w_kernel<<<tw_blocks, 256, 0, stream>>>(Wv, WvT, (int)K, (int)N);
    transpose_w_kernel<<<tw_blocks, 256, 0, stream>>>(Wo, WoT, (int)K, (int)N);
    // 3) QKV projections (WMMA, 16x64 tile per wave)
    const int gemm_blocks = (int)((M / 16) * (N / 64) / 8);
    gemm_bf16_kernel<<<gemm_blocks, 256, 0, stream>>>(Xb, WqT, bq, Qraw, (int)M, (int)N, (int)K);
    gemm_bf16_kernel<<<gemm_blocks, 256, 0, stream>>>(Xb, WkT, bk, Kraw, (int)M, (int)N, (int)K);
    gemm_bf16_kernel<<<gemm_blocks, 256, 0, stream>>>(Xb, WvT, bv, Vraw, (int)M, (int)N, (int)K);
    // 4) per-head LN + RoPE + pack
    norm_rope_kernel<<<(MROW * HH) / 8, 256, 0, stream>>>(Qraw, Kraw, Vraw, rcos, rsin,
                                                          qnw, qnb, knw, knb, Qn, Kn, VTb);
    // 5) flash attention (WMMA + TDM double buffering)
    flash_attn_kernel<<<(BB * HH * (NN / 16)) / 8, 256, 0, stream>>>(Qn, Kn, VTb, Obf);
    // 6) output projection (WMMA) -> d_out
    gemm_bf16_kernel<<<gemm_blocks, 256, 0, stream>>>(Obf, WoT, bo, out, (int)M, (int)DIM, (int)K);
    (void)in_sizes; (void)n_in; (void)out_size; (void)ws_size;
}
